// MultiheadInteraction_57260503990665
// MI455X (gfx1250) — matrix operand order repaired and checked by reading, not compiled, not run on hardware
//
#include <hip/hip_runtime.h>
#include <hip/hip_bf16.h>

typedef __attribute__((ext_vector_type(16))) _Float16 v16h;
typedef __attribute__((ext_vector_type(8)))  _Float16 v8h;
typedef __attribute__((ext_vector_type(4)))  _Float16 v4h;
typedef __attribute__((ext_vector_type(2)))  _Float16 v2h;
typedef __attribute__((ext_vector_type(8)))  float    v8f;
typedef __attribute__((ext_vector_type(4)))  float    v4f;

#define S_TOTAL 4096
#define D_TOTAL 1024
#define HEADS   16
#define HD      64      // per-head dim (d == e == 64)
#define CHUNK   512     // sequence rows per kv-reduction block
#define KSTEP   32      // WMMA K per step (f16 16x16x32)

// feat(x) = (tanh(x)+1)/2 -> native V_TANH_F32 + one mixed-precision FMA.
__device__ __forceinline__ float featf32(float x) {
#if __has_builtin(__builtin_amdgcn_tanhf)
    return __builtin_fmaf(__builtin_amdgcn_tanhf(x), 0.5f, 0.5f);
#elif __has_builtin(__builtin_amdgcn_tanh_f32)
    return __builtin_fmaf(__builtin_amdgcn_tanh_f32(x), 0.5f, 0.5f);
#else
    return __builtin_amdgcn_rcpf(1.0f +
           __builtin_amdgcn_exp2f(-2.8853900817779268f * x));
#endif
}
__device__ __forceinline__ _Float16 featf(float x) { return (_Float16)featf32(x); }

__device__ __forceinline__ v16h cat8(v8h lo, v8h hi) {
    return __builtin_shufflevector(lo, hi, 0,1,2,3,4,5,6,7,8,9,10,11,12,13,14,15);
}

__global__ void zero_ws_kernel(float* __restrict__ p, int n) {
    int i = blockIdx.x * 256 + threadIdx.x;
    if (i < n) p[i] = 0.0f;
}

// ---------------------------------------------------------------------------
// Kernel 1: kv[b,h][64][64] += feat(K)^T * V over a 512-row sequence chunk,
//           ksum[b,h][64]   += column sums of feat(K).
// grid = (64 bh, 8 chunks), block = 256 (8 waves, each wave owns 2 MxN tiles)
// ---------------------------------------------------------------------------
__global__ __launch_bounds__(256) void kv_kernel(const float* __restrict__ kin,
                                                 const float* __restrict__ vin,
                                                 float* __restrict__ kv_ws,
                                                 float* __restrict__ ksum_ws) {
    // transposed staging: [feature-column][sequence-row], stride 40 halfs (80B,
    // 16B-aligned rows, 20-dword stride -> conflict-free 16-lane column reads)
    __shared__ _Float16 ldsK[HD][40];
    __shared__ _Float16 ldsV[HD][40];

    const int tid  = threadIdx.x;
    const int bh   = blockIdx.x;            // 0..63
    const int b    = bh >> 4;
    const int h    = bh & 15;
    const int sbeg = blockIdx.y * CHUNK;

    const int lane = tid & 31;
    const int wave = tid >> 5;
    const int half = lane >> 4;             // 0 or 1
    const int l16  = lane & 15;

    const int mi    = wave & 3;             // A row-tile (d)
    const int nbase = (wave >> 2) * 2;      // two B col-tiles (e)

    // staging map: 2 adjacent rows x 4 cols per thread -> packed v2h LDS stores
    const int srow = (tid >> 4) * 2;        // 0,2,...,30
    const int col0 = (tid & 15) * 4;        // 0..60

    const size_t head_off = (size_t)b * S_TOTAL * D_TOTAL + (size_t)h * HD;
    const float* kbase = kin + head_off;
    const float* vbase = vin + head_off;

    v8f c0 = {}; v8f c1 = {};
    float ks_acc = 0.0f;

    for (int s0 = sbeg; s0 < sbeg + CHUNK; s0 += KSTEP) {
        const float* krow0 = kbase + (size_t)(s0 + srow) * D_TOTAL + col0;
        const float* vrow0 = vbase + (size_t)(s0 + srow) * D_TOTAL + col0;
        v4f ka = __builtin_nontemporal_load((const v4f*)(krow0));
        v4f kb = __builtin_nontemporal_load((const v4f*)(krow0 + D_TOTAL));
        v4f va = __builtin_nontemporal_load((const v4f*)(vrow0));
        v4f vb = __builtin_nontemporal_load((const v4f*)(vrow0 + D_TOTAL));

#pragma unroll
        for (int j = 0; j < 4; ++j) {
            v2h pk; pk[0] = featf(ka[j]);      pk[1] = featf(kb[j]);
            v2h pv; pv[0] = (_Float16)va[j];   pv[1] = (_Float16)vb[j];
            *(v2h*)&ldsK[col0 + j][srow] = pk;     // adjacent rows are contiguous
            *(v2h*)&ldsV[col0 + j][srow] = pv;
        }

        __syncthreads();

        // partial ksum: column t summed over the 32 staged rows (vector LDS reads)
        if (tid < HD) {
            const v8h* kp = (const v8h*)&ldsK[tid][0];
            float a = 0.0f;
#pragma unroll
            for (int c = 0; c < 4; ++c) {
                v8h x = kp[c];
#pragma unroll
                for (int j = 0; j < 8; ++j) a += (float)x[j];
            }
            ks_acc += a;
        }

        // A fragment (16x32 f16): lane half 0 -> K 0-7 & 16-23, half 1 -> K 8-15 & 24-31
        const int m = mi * 16 + l16;
        v8h alo = *(const v8h*)&ldsK[m][half * 8];
        v8h ahi = *(const v8h*)&ldsK[m][16 + half * 8];
        v16h afr = cat8(alo, ahi);

        // B fragments (32x16 f16): lane half 0 -> K 0-7 & 16-23, half 1 -> K 8-15 & 24-31, lane%16 = N
        const int n0 = nbase * 16 + l16;
        v8h b0lo = *(const v8h*)&ldsV[n0][half * 8];
        v8h b0hi = *(const v8h*)&ldsV[n0][16 + half * 8];
        v16h bf0 = cat8(b0lo, b0hi);
        v8h b1lo = *(const v8h*)&ldsV[n0 + 16][half * 8];
        v8h b1hi = *(const v8h*)&ldsV[n0 + 16][16 + half * 8];
        v16h bf1 = cat8(b1lo, b1hi);

        c0 = __builtin_amdgcn_wmma_f32_16x16x32_f16(false, afr, false, bf0,
                                                    (short)0, c0, false, false);
        c1 = __builtin_amdgcn_wmma_f32_16x16x32_f16(false, afr, false, bf1,
                                                    (short)0, c1, false, false);
        __syncthreads();
    }

    // accumulate partial results to workspace
    float* kvout = kv_ws + (size_t)bh * HD * HD;
#pragma unroll
    for (int r = 0; r < 8; ++r) {
        const int m = mi * 16 + r + 8 * half;   // C layout: vgpr r, lane half
        atomicAdd(&kvout[m * HD + nbase * 16 + l16],       c0[r]);
        atomicAdd(&kvout[m * HD + (nbase + 1) * 16 + l16], c1[r]);
    }
    if (tid < HD) atomicAdd(&ksum_ws[bh * HD + tid], ks_acc);
}

// ---------------------------------------------------------------------------
// Kernel 2: out[s,:] = (feat(Q[s,:]) @ kv) * 1/(feat(Q[s,:]) . (ksum+1e-8))
// grid = (64 bh, 64 S-tiles of 64 rows), block = 256 (8 waves x 2 tiles)
// ---------------------------------------------------------------------------
__global__ __launch_bounds__(256) void out_kernel(const float* __restrict__ qin,
                                                  const float* __restrict__ kv_ws,
                                                  const float* __restrict__ ksum_ws,
                                                  float* __restrict__ out) {
    __shared__ _Float16 ldsKV[HD][72];   // [n][k], stride 72 halfs (144B)
    __shared__ _Float16 ldsQ[HD][72];    // [m][k]
    __shared__ float    ldsKs[HD];
    __shared__ float    ldsZ[HD];

    const int tid   = threadIdx.x;
    const int bh    = blockIdx.x;
    const int b     = bh >> 4;
    const int h     = bh & 15;
    const int sbase = blockIdx.y * 64;

    // stage kv (f32 -> f16, transposed to [n][k]): thread owns (n, 4 consecutive k)
    // -> coalesced global reads (64 lanes x consecutive n), packed v4h b64 stores
    const float* kvsrc = kv_ws + (size_t)bh * HD * HD;
    {
        const int n  = tid & 63;
        const int kq = (tid >> 6) * 4;      // 0,4,8,12
#pragma unroll
        for (int g = 0; g < 4; ++g) {
            const int kk0 = g * 16 + kq;
            v4h p;
#pragma unroll
            for (int j = 0; j < 4; ++j)
                p[j] = (_Float16)kvsrc[(kk0 + j) * HD + n];
            *(v4h*)&ldsKV[n][kk0] = p;
        }
    }
    if (tid < HD) ldsKs[tid] = ksum_ws[bh * HD + tid] + 1e-8f;

    // stage feat(Q): 64 rows x 64 cols, 16 floats per thread, v8h b128 stores
    {
        const int row = tid >> 2;
        const int c0  = (tid & 3) * 16;
        const float* qrow = qin + ((size_t)b * S_TOTAL + sbase + row) * D_TOTAL
                                + (size_t)h * HD + c0;
#pragma unroll
        for (int g = 0; g < 2; ++g) {
            v4f f0 = __builtin_nontemporal_load((const v4f*)(qrow + g * 8));
            v4f f1 = __builtin_nontemporal_load((const v4f*)(qrow + g * 8 + 4));
            v8h p;
#pragma unroll
            for (int e = 0; e < 4; ++e) { p[e] = featf(f0[e]); p[4 + e] = featf(f1[e]); }
            *(v8h*)&ldsQ[row][c0 + g * 8] = p;
        }
    }
    __syncthreads();

    // per-row normalizer z = 1 / (qh . k_reduced), vector LDS reads
    if (tid < HD) {
        const v8h* qp  = (const v8h*)&ldsQ[tid][0];
        const v4f* ksp = (const v4f*)ldsKs;
        float dot = 0.0f;
#pragma unroll
        for (int c = 0; c < 8; ++c) {
            v8h x  = qp[c];
            v4f k0 = ksp[2 * c];
            v4f k1 = ksp[2 * c + 1];
#pragma unroll
            for (int j = 0; j < 4; ++j) {
                dot += (float)x[j]     * k0[j];
                dot += (float)x[4 + j] * k1[j];
            }
        }
        ldsZ[tid] = __builtin_amdgcn_rcpf(dot);
    }
    __syncthreads();

    const int lane = tid & 31;
    const int wave = tid >> 5;
    const int half = lane >> 4;
    const int l16  = lane & 15;
    const int mi    = wave & 3;
    const int nbase = (wave >> 2) * 2;

    v8f c0 = {}; v8f c1 = {};
#pragma unroll
    for (int ks = 0; ks < 2; ++ks) {        // K = 64 in two 32-steps
        const int m = mi * 16 + l16;
        v8h alo = *(const v8h*)&ldsQ[m][ks * 32 + half * 8];
        v8h ahi = *(const v8h*)&ldsQ[m][ks * 32 + 16 + half * 8];
        v16h afr = cat8(alo, ahi);

        const int n0 = nbase * 16 + l16;
        v8h b0lo = *(const v8h*)&ldsKV[n0][ks * 32 + half * 8];
        v8h b0hi = *(const v8h*)&ldsKV[n0][ks * 32 + 16 + half * 8];
        v16h bf0 = cat8(b0lo, b0hi);
        v8h b1lo = *(const v8h*)&ldsKV[n0 + 16][ks * 32 + half * 8];
        v8h b1hi = *(const v8h*)&ldsKV[n0 + 16][ks * 32 + 16 + half * 8];
        v16h bf1 = cat8(b1lo, b1hi);

        c0 = __builtin_amdgcn_wmma_f32_16x16x32_f16(false, afr, false, bf0,
                                                    (short)0, c0, false, false);
        c1 = __builtin_amdgcn_wmma_f32_16x16x32_f16(false, afr, false, bf1,
                                                    (short)0, c1, false, false);
    }

    // scale by z and stream results out (write-once -> non-temporal stores)
    float* obase = out + ((size_t)b * S_TOTAL + sbase) * D_TOTAL + (size_t)h * HD;
#pragma unroll
    for (int r = 0; r < 8; ++r) {
        const int m  = mi * 16 + r + 8 * half;
        const float z = ldsZ[m];
        __builtin_nontemporal_store(c0[r] * z,
            &obase[(size_t)m * D_TOTAL + nbase * 16 + l16]);
        __builtin_nontemporal_store(c1[r] * z,
            &obase[(size_t)m * D_TOTAL + (nbase + 1) * 16 + l16]);
    }
}

extern "C" void kernel_launch(void* const* d_in, const int* in_sizes, int n_in,
                              void* d_out, int out_size, void* d_ws, size_t ws_size,
                              hipStream_t stream) {
    const float* q = (const float*)d_in[0];
    const float* k = (const float*)d_in[1];
    const float* v = (const float*)d_in[2];
    float* out = (float*)d_out;

    float* kv_ws   = (float*)d_ws;                        // 64 * 64 * 64 f32 = 1 MB
    float* ksum_ws = kv_ws + (size_t)64 * HD * HD;        // 64 * 64 f32

    const int zn = 64 * HD * HD + 64 * HD;
    zero_ws_kernel<<<(zn + 255) / 256, 256, 0, stream>>>(kv_ws, zn);

    dim3 g1(64, S_TOTAL / CHUNK);   // 64 (b,h) x 8 chunks
    kv_kernel<<<g1, 256, 0, stream>>>(k, v, kv_ws, ksum_ws);

    dim3 g2(64, S_TOTAL / 64);      // 64 (b,h) x 64 row-tiles
    out_kernel<<<g2, 256, 0, stream>>>(q, kv_ws, ksum_ws, out);
}
